// MedicalMoE_19868518711317
// MI455X (gfx1250) — compile-verified, matrix operand order checked by reading, not run on hardware
//
#include <hip/hip_runtime.h>
#include <hip/hip_bf16.h>

// ---------------- problem constants ----------------
#define H_DIM 768
#define S_LEN 512
#define B_SZ  2
#define N_TOK 1024           // B*S
#define NHEAD 8
#define HDDIM 96             // H/NH
#define NEXP  12
#define NSPEC 12
#define KMAX5 5

typedef __attribute__((ext_vector_type(16))) __bf16       v16bf;
typedef __attribute__((ext_vector_type(8)))  float        v8f;
typedef __attribute__((ext_vector_type(8)))  unsigned int v8u;

__device__ __forceinline__ unsigned short f2bf(float f) {
  unsigned u = __float_as_uint(f);
  u += 0x7FFFu + ((u >> 16) & 1u);      // round-to-nearest-even
  return (unsigned short)(u >> 16);
}
__device__ __forceinline__ unsigned pack2bf(float a, float b) {
  return (unsigned)f2bf(a) | ((unsigned)f2bf(b) << 16);
}
__device__ __forceinline__ float gelu_exact(float x) {
  return 0.5f * x * (1.0f + erff(x * 0.70710678118654752f));
}

// ---------------------------------------------------------------------------
// Generic bf16-WMMA GEMM: C = act( scale * (A @ B) + bias )
// A: [M,K] row-major (lda), B: [K,N] row-major (ldb) or [N,K] if TRANSB,
// C: [M,N] (ldc). Batched via grid.z with (b,h) = (z/nh, z%nh) strides.
// Block: 128 threads (4 waves), tile 64x64, BK=32. Each wave: 2x2 WMMA tiles.
// Both operands use identical pair-packed LDS layout (stride 18 dwords):
//   As2[r*18 + k/2]  : row-major rows, bf16 K-pairs per dword
//   Bs2[n*18 + k/2]  : column-major cols, bf16 K-pairs per dword
// -> each lane's fragment = two contiguous 4-dword runs (ds_load_2addr_b64,
//    no register repacking in front of v_wmma).
// FULL=1: M%64==0, N%64==0, K%32==0 (straight-line vectorized loads).
// ---------------------------------------------------------------------------
template <int TRANSB, int FULL>
__global__ __launch_bounds__(128) void gemm_wmma_bf16(
    const float* __restrict__ A, const float* __restrict__ B,
    const float* __restrict__ bias, float* __restrict__ C,
    int M, int N, int K, int lda, int ldb, int ldc,
    int act, float scale, int nh,
    long long sAb, long long sAh, long long sBb, long long sBh,
    long long sCb, long long sCh)
{
  __shared__ __align__(16) unsigned As2[64 * 18];   // 4608 B
  __shared__ __align__(16) unsigned Bs2[64 * 18];   // 4608 B

  int z = blockIdx.z, bz = z / nh, hz = z - bz * nh;
  A += bz * sAb + hz * sAh;
  B += bz * sBb + hz * sBh;
  C += bz * sCb + hz * sCh;

  const int m0 = blockIdx.y * 64, n0 = blockIdx.x * 64;
  const int tid = threadIdx.x, lane = tid & 31, wave = tid >> 5;
  const int wr = (wave >> 1) * 32, wc = (wave & 1) * 32;
  const int lrow = lane & 15;
  const int kc = (lane & 16) ? 4 : 0;      // packed-pair offset (ISA: hi-lanes hold K+8)
  v8f acc[2][2] = {};

  for (int kk = 0; kk < K; kk += 32) {
    if (FULL) {
      // ---------- fast path: b128 global loads, packed LDS stores ----------
      if (kk + 32 < K) {  // prefetch next K-slab (global_prefetch_b8)
        __builtin_prefetch(A + (long long)(m0 + (tid & 63)) * lda + kk + 32, 0, 1);
        if (TRANSB)
          __builtin_prefetch(B + (long long)(n0 + (tid & 63)) * ldb + kk + 32, 0, 1);
        else
          __builtin_prefetch(B + (long long)(kk + 32 + (tid & 31)) * ldb + n0, 0, 1);
      }
#pragma unroll
      for (int i = 0; i < 4; i++) {        // A: 64 rows x 8 k-quads
        int idx = i * 128 + tid, r = idx >> 3, cb = (idx & 7) * 4;
        float4 v = *(const float4*)(A + (long long)(m0 + r) * lda + kk + cb);
        uint2 pk;
        pk.x = pack2bf(v.x, v.y);
        pk.y = pack2bf(v.z, v.w);
        *(uint2*)&As2[r * 18 + (cb >> 1)] = pk;
      }
      if (TRANSB) {
#pragma unroll
        for (int i = 0; i < 4; i++) {      // B[N,K]: 64 cols x 8 k-quads
          int idx = i * 128 + tid, n = idx >> 3, cb = (idx & 7) * 4;
          float4 v = *(const float4*)(B + (long long)(n0 + n) * ldb + kk + cb);
          uint2 pk;
          pk.x = pack2bf(v.x, v.y);
          pk.y = pack2bf(v.z, v.w);
          *(uint2*)&Bs2[n * 18 + (cb >> 1)] = pk;
        }
      } else {
#pragma unroll
        for (int i = 0; i < 2; i++) {      // B[K,N]: 16 k-pairs x 16 n-quads
          int idx = i * 128 + tid, kp = idx >> 4, nb = (idx & 15) * 4;
          const float* r0 = B + (long long)(kk + 2 * kp) * ldb + n0 + nb;
          float4 v0 = *(const float4*)r0;
          float4 v1 = *(const float4*)(r0 + ldb);
          Bs2[(nb + 0) * 18 + kp] = pack2bf(v0.x, v1.x);
          Bs2[(nb + 1) * 18 + kp] = pack2bf(v0.y, v1.y);
          Bs2[(nb + 2) * 18 + kp] = pack2bf(v0.z, v1.z);
          Bs2[(nb + 3) * 18 + kp] = pack2bf(v0.w, v1.w);
        }
      }
    } else {
      // ---------- edge path: guarded scalar loads into the same packed layouts ----------
      unsigned short* Ah = (unsigned short*)As2;
      unsigned short* Bh = (unsigned short*)Bs2;
#pragma unroll
      for (int i = 0; i < 16; i++) {
        int idx = i * 128 + tid, r = idx >> 5, c = idx & 31;
        int gm = m0 + r, gk = kk + c;
        Ah[(r * 18 + (c >> 1)) * 2 + (c & 1)] =
            f2bf((gm < M && gk < K) ? A[(long long)gm * lda + gk] : 0.0f);
      }
#pragma unroll
      for (int i = 0; i < 16; i++) {
        int idx = i * 128 + tid, n = idx >> 5, c = idx & 31;
        int gn = n0 + n, gk = kk + c;
        float v = 0.0f;
        if (gn < N && gk < K)
          v = TRANSB ? B[(long long)gn * ldb + gk] : B[(long long)gk * ldb + gn];
        Bh[(n * 18 + (c >> 1)) * 2 + (c & 1)] = f2bf(v);
      }
    }
    __syncthreads();

    // --- fragments: two contiguous 4-dword runs per lane (pair idx = kh>>1) ---
    v16bf af[2], bfm[2];
#pragma unroll
    for (int tm = 0; tm < 2; tm++) {
      const unsigned* base = &As2[(wr + tm * 16 + lrow) * 18 + kc];
      v8u t;
#pragma unroll
      for (int v = 0; v < 8; v++) t[v] = base[v + ((v < 4) ? 0 : 4)];
      af[tm] = __builtin_bit_cast(v16bf, t);
    }
#pragma unroll
    for (int tn = 0; tn < 2; tn++) {
      const unsigned* base = &Bs2[(wc + tn * 16 + lrow) * 18 + kc];
      v8u t;
#pragma unroll
      for (int v = 0; v < 8; v++) t[v] = base[v + ((v < 4) ? 0 : 4)];
      bfm[tn] = __builtin_bit_cast(v16bf, t);
    }

#pragma unroll
    for (int tm = 0; tm < 2; tm++)
#pragma unroll
      for (int tn = 0; tn < 2; tn++)
        acc[tm][tn] = __builtin_amdgcn_wmma_f32_16x16x32_bf16(
            false, af[tm], false, bfm[tn], (short)0, acc[tm][tn], false, false);
    __syncthreads();
  }

  // --- epilogue: scale, bias, activation, store (C layout: VGPR v -> M, lane -> N) ---
  const int mb = (lane & 16) ? 8 : 0;
#pragma unroll
  for (int tm = 0; tm < 2; tm++)
#pragma unroll
    for (int tn = 0; tn < 2; tn++)
#pragma unroll
      for (int v = 0; v < 8; v++) {
        int mrow = m0 + wr + tm * 16 + mb + v;
        int ncol = n0 + wc + tn * 16 + lrow;
        if (FULL || (mrow < M && ncol < N)) {
          float val = acc[tm][tn][v] * scale;
          if (bias) val += bias[ncol];
          if (act == 1)      val = gelu_exact(val);
          else if (act == 2) val = fmaxf(val, 0.0f);
          C[(long long)mrow * ldc + ncol] = val;
        }
      }
}

// ---------------- support kernels ----------------

__global__ void softmax_rows(float* __restrict__ X, int L) {
  float* r = X + (long long)blockIdx.x * L;
  __shared__ float red[128];
  int t = threadIdx.x;
  float m = -1e30f;
  for (int j = t; j < L; j += 128) m = fmaxf(m, r[j]);
  red[t] = m; __syncthreads();
  for (int o = 64; o > 0; o >>= 1) { if (t < o) red[t] = fmaxf(red[t], red[t + o]); __syncthreads(); }
  m = red[0]; __syncthreads();
  float s = 0.0f;
  for (int j = t; j < L; j += 128) { float e = expf(r[j] - m); r[j] = e; s += e; }
  red[t] = s; __syncthreads();
  for (int o = 64; o > 0; o >>= 1) { if (t < o) red[t] += red[t + o]; __syncthreads(); }
  float inv = 1.0f / red[0];
  for (int j = t; j < L; j += 128) r[j] *= inv;
}

__global__ void layernorm_rows(float* __restrict__ X, const float* __restrict__ g,
                               const float* __restrict__ b, int Hd) {
  float* r = X + (long long)blockIdx.x * Hd;
  __shared__ float red[256];
  int t = threadIdx.x;
  float s = 0.0f;
  for (int j = t; j < Hd; j += 256) s += r[j];
  red[t] = s; __syncthreads();
  for (int o = 128; o > 0; o >>= 1) { if (t < o) red[t] += red[t + o]; __syncthreads(); }
  float mu = red[0] / Hd; __syncthreads();
  float v = 0.0f;
  for (int j = t; j < Hd; j += 256) { float d = r[j] - mu; v += d * d; }
  red[t] = v; __syncthreads();
  for (int o = 128; o > 0; o >>= 1) { if (t < o) red[t] += red[t + o]; __syncthreads(); }
  float inv = rsqrtf(red[0] / Hd + 1e-5f);
  for (int j = t; j < Hd; j += 256) r[j] = (r[j] - mu) * inv * g[j] + b[j];
}

__global__ void meanpool_seq(const float* __restrict__ X, float* __restrict__ P, int S, int Hd) {
  int b = blockIdx.x, h = threadIdx.x;
  if (h >= Hd) return;
  float s = 0.0f;
  for (int j = 0; j < S; j++) s += X[((long long)b * S + j) * Hd + h];
  P[b * Hd + h] = s / (float)S;
}

// tiny MLP layer for M=2 rows: act 0=none,1=relu,2=sigmoid
__global__ void small_mlp(const float* __restrict__ A, const float* __restrict__ W,
                          const float* __restrict__ bias, float* __restrict__ Y,
                          int K, int N, int act) {
  int m = blockIdx.x, n = threadIdx.x;
  if (n >= N) return;
  float s = bias ? bias[n] : 0.0f;
  const float* a = A + (long long)m * K;
  for (int k = 0; k < K; k++) s += a[k] * W[(long long)k * N + n];
  if (act == 1)      s = fmaxf(s, 0.0f);
  else if (act == 2) s = 1.0f / (1.0f + expf(-s));
  Y[(long long)m * N + n] = s;
}

__global__ void spec_finalize(const float* __restrict__ spec_logits,
                              const float* __restrict__ pooled,
                              float* __restrict__ feats, float* __restrict__ dout_spec) {
  int b = threadIdx.x;
  if (b >= B_SZ) return;
  float pr[NSPEC], mx = -1e30f, s = 0.0f;
  for (int e = 0; e < NSPEC; e++) mx = fmaxf(mx, spec_logits[b * NSPEC + e]);
  for (int e = 0; e < NSPEC; e++) { pr[e] = expf(spec_logits[b * NSPEC + e] - mx); s += pr[e]; }
  for (int e = 0; e < NSPEC; e++) {
    pr[e] /= s;
    dout_spec[b * NSPEC + e] = pr[e];
    feats[b * (H_DIM + NSPEC) + H_DIM + e] = pr[e];
  }
  for (int j = 0; j < H_DIM; j++) feats[b * (H_DIM + NSPEC) + j] = pooled[b * H_DIM + j];
}

__global__ void router_topk(const float* __restrict__ logits, const float* __restrict__ urg,
                            float* __restrict__ gate, float* __restrict__ cnt,
                            float* __restrict__ dout_urg, float* __restrict__ dout_usage) {
  __shared__ float sidx[B_SZ * KMAX5];
  if (threadIdx.x == 0) {
    float um = 0.5f * (urg[0] + urg[1]);
    int kact = (um > 0.7f) ? KMAX5 : 3;
    for (int b = 0; b < B_SZ; b++) {
      float pr[NEXP], mx = -1e30f, s = 0.0f;
      for (int e = 0; e < NEXP; e++) mx = fmaxf(mx, logits[b * NEXP + e]);
      for (int e = 0; e < NEXP; e++) { pr[e] = expf(logits[b * NEXP + e] - mx); s += pr[e]; }
      for (int e = 0; e < NEXP; e++) pr[e] /= s;
      int id[KMAX5]; float w[KMAX5]; bool used[NEXP] = {};
      for (int slot = 0; slot < KMAX5; slot++) {            // jax top_k tie-break: first index
        int best = -1; float bv = -1e30f;
        for (int e = 0; e < NEXP; e++)
          if (!used[e] && pr[e] > bv) { bv = pr[e]; best = e; }
        used[best] = true; id[slot] = best; w[slot] = bv;
      }
      float mx2 = -1e30f, ss = 0.0f, sw[KMAX5];
      for (int slot = 0; slot < kact; slot++) mx2 = fmaxf(mx2, w[slot]);
      for (int slot = 0; slot < KMAX5; slot++) {
        if (slot < kact) { sw[slot] = expf(w[slot] - mx2); ss += sw[slot]; }
        else sw[slot] = 0.0f;
      }
      for (int slot = 0; slot < kact; slot++) sw[slot] /= ss;
      for (int e = 0; e < NEXP; e++) { gate[b * NEXP + e] = 0.0f; cnt[b * NEXP + e] = 0.0f; }
      for (int slot = 0; slot < kact; slot++) {
        gate[b * NEXP + id[slot]] += sw[slot];
        cnt[b * NEXP + id[slot]]  += 1.0f;
      }
      for (int slot = 0; slot < KMAX5; slot++) sidx[b * KMAX5 + slot] = (float)id[slot];
      dout_urg[b] = urg[b];
    }
  }
  __syncthreads();
  for (int i = threadIdx.x; i < B_SZ * S_LEN * KMAX5; i += blockDim.x) {
    int b = i / (S_LEN * KMAX5), slot = i % KMAX5;
    dout_usage[i] = sidx[b * KMAX5 + slot];
  }
}

__global__ void conf_head(const float* __restrict__ Hd, const float* __restrict__ w,
                          const float* __restrict__ b2, float* __restrict__ conf) {
  int n = blockIdx.x * blockDim.x + threadIdx.x;
  if (n >= N_TOK) return;
  const float* r = Hd + (long long)n * 192;
  float s = b2[0];
  for (int i = 0; i < 192; i++) s += r[i] * w[i];
  conf[n] = 1.0f / (1.0f + expf(-s));
}

__global__ void accum_expert(const float* __restrict__ he, const float* __restrict__ conf,
                             const float* __restrict__ gate, const float* __restrict__ cnt,
                             float* __restrict__ out, float* __restrict__ tconf, int e) {
  int i = blockIdx.x * blockDim.x + threadIdx.x;
  if (i >= N_TOK * H_DIM) return;
  int n = i / H_DIM, b = n >> 9;             // S=512
  float c = conf[n];
  out[i] += gate[b * NEXP + e] * c * he[i];
  if (i - n * H_DIM == 0) tconf[n] += cnt[b * NEXP + e] * c;
}

__global__ void finalize_output(const float* __restrict__ acc, const float* __restrict__ safe,
                                float* __restrict__ o_out, float* __restrict__ o_safe) {
  int i = blockIdx.x * blockDim.x + threadIdx.x;
  if (i < B_SZ) o_safe[i] = safe[i];
  if (i >= N_TOK * H_DIM) return;
  int b = i / (S_LEN * H_DIM);
  float f = (safe[b] < 0.5f) ? 0.5f : 1.0f;
  o_out[i] = acc[i] * f;
}

__global__ void avg_conf_kernel(const float* __restrict__ tconf, float* __restrict__ out) {
  __shared__ float red[256];
  int t = threadIdx.x;
  float s = 0.0f;
  for (int j = t; j < N_TOK; j += 256) s += tconf[j];
  red[t] = s; __syncthreads();
  for (int o = 128; o > 0; o >>= 1) { if (t < o) red[t] += red[t + o]; __syncthreads(); }
  if (t == 0) out[0] = red[0] / (float)N_TOK;
}

// ---------------- host orchestration ----------------

static inline void launch_gemm(hipStream_t st, const float* A, const float* B, const float* bias,
    float* C, int M, int N, int K, int lda, int ldb, int ldc, int transB, int act, float scale,
    int batch = 1, int nh = 1,
    long long sAb = 0, long long sAh = 0, long long sBb = 0, long long sBh = 0,
    long long sCb = 0, long long sCh = 0)
{
  dim3 g((N + 63) / 64, (M + 63) / 64, batch), b(128, 1, 1);
  bool full = (M % 64 == 0) && (N % 64 == 0) && (K % 32 == 0);
  if (transB) {
    if (full) gemm_wmma_bf16<1, 1><<<g, b, 0, st>>>(A, B, bias, C, M, N, K, lda, ldb, ldc,
                                                    act, scale, nh, sAb, sAh, sBb, sBh, sCb, sCh);
    else      gemm_wmma_bf16<1, 0><<<g, b, 0, st>>>(A, B, bias, C, M, N, K, lda, ldb, ldc,
                                                    act, scale, nh, sAb, sAh, sBb, sBh, sCb, sCh);
  } else {
    if (full) gemm_wmma_bf16<0, 1><<<g, b, 0, st>>>(A, B, bias, C, M, N, K, lda, ldb, ldc,
                                                    act, scale, nh, sAb, sAh, sBb, sBh, sCb, sCh);
    else      gemm_wmma_bf16<0, 0><<<g, b, 0, st>>>(A, B, bias, C, M, N, K, lda, ldb, ldc,
                                                    act, scale, nh, sAb, sAh, sBb, sBh, sCb, sCh);
  }
}

// workspace layout (float elements)
#define OFF_QKV    0LL            // 1024*2304
#define OFF_ATTN   2359296LL      // 1024*768
#define OFF_ATTNP  3145728LL      // 1024*768
#define OFF_SMALL  3932160LL      // 16384 scratch for router
#define OFF_BUFA   3948544LL      // 1024*6144 (scores [16,512,512] aliased here)
#define OFF_BUFB   10240000LL     // 1024*3072
#define OFF_BUFC   13385728LL     // 1024*768
#define OFF_BUFD   14172160LL     // 1024*192
#define OFF_CONF   14368768LL     // 1024
#define OFF_OUT    14369792LL     // 1024*768
#define OFF_TCONF  15156224LL     // 1024
#define OFF_POOL2  15157248LL     // 1536
#define WS_FLOATS  15158784LL

// small-region sub-offsets
#define SM_POOLED 0
#define SM_SPECH  1536
#define SM_SPEC   3072
#define SM_FEATS  3104
#define SM_ROUTEH 4672
#define SM_LOGITS 5440
#define SM_URGH   5472
#define SM_URG    5856
#define SM_GATE   5888
#define SM_CNT    5920
#define SM_SAFH   5984
#define SM_SAFE   6368

// output offsets (tuple concatenated flat)
#define O_OUT    0LL
#define O_SPEC   786432LL
#define O_URG    786456LL
#define O_AVG    786458LL
#define O_SAFE   786459LL
#define O_USAGE  786461LL
#define O_EVID   791581LL

extern "C" void kernel_launch(void* const* d_in, const int* in_sizes, int n_in,
                              void* d_out, int out_size, void* d_ws, size_t ws_size,
                              hipStream_t stream) {
  if (ws_size < (size_t)WS_FLOATS * sizeof(float)) return;  // need ~61 MB scratch

  // ---- decode inputs (setup_inputs dict insertion order) ----
  int p = 0;
  const float* x     = (const float*)d_in[p++];
  const float* Wqkv  = (const float*)d_in[p++]; const float* bqkv = (const float*)d_in[p++];
  const float* Wo    = (const float*)d_in[p++]; const float* bo   = (const float*)d_in[p++];
  const float* cnW1  = (const float*)d_in[p++]; const float* cnb1 = (const float*)d_in[p++];
  const float* cnW2  = (const float*)d_in[p++]; const float* cnb2 = (const float*)d_in[p++];
  const float* rW1   = (const float*)d_in[p++]; const float* rb1  = (const float*)d_in[p++];
  const float* rW2   = (const float*)d_in[p++]; const float* rb2  = (const float*)d_in[p++];
  const float* uW1   = (const float*)d_in[p++]; const float* ub1  = (const float*)d_in[p++];
  const float* uW2   = (const float*)d_in[p++]; const float* ub2  = (const float*)d_in[p++];
  const float* sW1   = (const float*)d_in[p++]; const float* sb1  = (const float*)d_in[p++];
  const float* sW2   = (const float*)d_in[p++]; const float* sb2  = (const float*)d_in[p++];
  const float* eW1   = (const float*)d_in[p++]; const float* eb1  = (const float*)d_in[p++];
  const float* eW2   = (const float*)d_in[p++]; const float* eb2  = (const float*)d_in[p++];

  struct ExpertP {
    const float* W[4]; const float* bb[4];
    const float *lng, *lnb, *cw1, *cb1, *cw2, *cb2;
    int nl; int d[5];
  } ex[NEXP];
  for (int e = 0; e < NEXP; e++) {
    if (e == 3 || e == 4) { ex[e].nl = 3; int dd[4] = {768, 6144, 3072, 768};
      for (int i = 0; i < 4; i++) ex[e].d[i] = dd[i]; }
    else if (e == 5)      { ex[e].nl = 4; int dd[5] = {768, 3072, 3072, 1536, 768};
      for (int i = 0; i < 5; i++) ex[e].d[i] = dd[i]; }
    else                  { ex[e].nl = 3; int dd[4] = {768, 3072, 1536, 768};
      for (int i = 0; i < 4; i++) ex[e].d[i] = dd[i]; }
    for (int l = 0; l < ex[e].nl; l++) ex[e].W[l]  = (const float*)d_in[p++];
    for (int l = 0; l < ex[e].nl; l++) ex[e].bb[l] = (const float*)d_in[p++];
    ex[e].lng = (const float*)d_in[p++]; ex[e].lnb = (const float*)d_in[p++];
    ex[e].cw1 = (const float*)d_in[p++]; ex[e].cb1 = (const float*)d_in[p++];
    ex[e].cw2 = (const float*)d_in[p++]; ex[e].cb2 = (const float*)d_in[p++];
  }

  float* ws   = (float*)d_ws;
  float* dout = (float*)d_out;
  float* qkvB   = ws + OFF_QKV;
  float* attnB  = ws + OFF_ATTN;
  float* attnP  = ws + OFF_ATTNP;
  float* sm     = ws + OFF_SMALL;
  float* bufA   = ws + OFF_BUFA;      // also scores
  float* bufB   = ws + OFF_BUFB;
  float* bufC   = ws + OFF_BUFC;
  float* bufD   = ws + OFF_BUFD;
  float* confB  = ws + OFF_CONF;
  float* outAcc = ws + OFF_OUT;
  float* tconf  = ws + OFF_TCONF;
  float* pool2  = ws + OFF_POOL2;

  const long long TOK3H = (long long)S_LEN * 3 * H_DIM;   // per-batch stride in qkv

  // ---- MHA ----
  launch_gemm(stream, x, Wqkv, bqkv, qkvB, N_TOK, 3 * H_DIM, H_DIM, H_DIM, 3 * H_DIM, 3 * H_DIM, 0, 0, 1.0f);
  // scores[b,h] = Q @ K^T / sqrt(hd)  (Q at col h*96, K at 768+h*96)
  launch_gemm(stream, qkvB, qkvB + H_DIM, nullptr, bufA,
              S_LEN, S_LEN, HDDIM, 3 * H_DIM, 3 * H_DIM, S_LEN, /*transB=*/1, 0,
              0.10206207261596576f /* 1/sqrt(96) */,
              B_SZ * NHEAD, NHEAD, TOK3H, HDDIM, TOK3H, HDDIM,
              (long long)NHEAD * S_LEN * S_LEN, (long long)S_LEN * S_LEN);
  softmax_rows<<<B_SZ * NHEAD * S_LEN, 128, 0, stream>>>(bufA, S_LEN);
  // attn[b,h] = P @ V  -> packed heads into attnB [N,768]
  launch_gemm(stream, bufA, qkvB + 2 * H_DIM, nullptr, attnB,
              S_LEN, HDDIM, S_LEN, S_LEN, 3 * H_DIM, H_DIM, 0, 0, 1.0f,
              B_SZ * NHEAD, NHEAD,
              (long long)NHEAD * S_LEN * S_LEN, (long long)S_LEN * S_LEN,
              TOK3H, HDDIM, (long long)S_LEN * H_DIM, HDDIM);
  launch_gemm(stream, attnB, Wo, bo, attnP, N_TOK, H_DIM, H_DIM, H_DIM, H_DIM, H_DIM, 0, 0, 1.0f);
  meanpool_seq<<<B_SZ, H_DIM, 0, stream>>>(attnP, sm + SM_POOLED, S_LEN, H_DIM);

  // ---- router (tiny MLPs, M=2) ----
  small_mlp<<<B_SZ, H_DIM, 0, stream>>>(sm + SM_POOLED, cnW1, cnb1, sm + SM_SPECH, H_DIM, H_DIM, 1);
  small_mlp<<<B_SZ, NSPEC, 0, stream>>>(sm + SM_SPECH, cnW2, cnb2, sm + SM_SPEC, H_DIM, NSPEC, 0);
  spec_finalize<<<1, 32, 0, stream>>>(sm + SM_SPEC, sm + SM_POOLED, sm + SM_FEATS, dout + O_SPEC);
  small_mlp<<<B_SZ, 384, 0, stream>>>(sm + SM_FEATS, rW1, rb1, sm + SM_ROUTEH, H_DIM + NSPEC, 384, 1);
  small_mlp<<<B_SZ, NEXP, 0, stream>>>(sm + SM_ROUTEH, rW2, rb2, sm + SM_LOGITS, 384, NEXP, 0);
  small_mlp<<<B_SZ, 192, 0, stream>>>(sm + SM_POOLED, uW1, ub1, sm + SM_URGH, H_DIM, 192, 1);
  small_mlp<<<B_SZ, 1, 0, stream>>>(sm + SM_URGH, uW2, ub2, sm + SM_URG, 192, 1, 2);
  router_topk<<<1, 64, 0, stream>>>(sm + SM_LOGITS, sm + SM_URG, sm + SM_GATE, sm + SM_CNT,
                                    dout + O_URG, dout + O_USAGE);

  // zero accumulators (OUT and TCONF are adjacent)
  hipMemsetAsync(outAcc, 0, (size_t)(N_TOK * H_DIM + N_TOK) * sizeof(float), stream);

  // ---- experts (dense over all tokens; gate/conf-weighted accumulate) ----
  for (int e = 0; e < NEXP; e++) {
    float* dst[4];
    if (ex[e].nl == 3) { dst[0] = bufA; dst[1] = bufB; dst[2] = bufC; }
    else               { dst[0] = bufA; dst[1] = bufB; dst[2] = bufA; dst[3] = bufC; }
    const float* cur = x;
    for (int l = 0; l < ex[e].nl; l++) {
      int din = ex[e].d[l], dn = ex[e].d[l + 1];
      int act = (l == ex[e].nl - 1) ? 0 : 1;   // gelu on hidden layers
      launch_gemm(stream, cur, ex[e].W[l], ex[e].bb[l], dst[l],
                  N_TOK, dn, din, din, dn, dn, 0, act, 1.0f);
      cur = dst[l];
    }
    layernorm_rows<<<N_TOK, 256, 0, stream>>>(bufC, ex[e].lng, ex[e].lnb, H_DIM);
    launch_gemm(stream, bufC, ex[e].cw1, ex[e].cb1, bufD,
                N_TOK, 192, H_DIM, H_DIM, 192, 192, 0, 2, 1.0f);   // relu
    conf_head<<<(N_TOK + 255) / 256, 256, 0, stream>>>(bufD, ex[e].cw2, ex[e].cb2, confB);
    accum_expert<<<(N_TOK * H_DIM + 255) / 256, 256, 0, stream>>>(
        bufC, confB, sm + SM_GATE, sm + SM_CNT, outAcc, tconf, e);
  }

  // ---- safety gate + outputs ----
  meanpool_seq<<<B_SZ, H_DIM, 0, stream>>>(outAcc, pool2, S_LEN, H_DIM);
  small_mlp<<<B_SZ, 384, 0, stream>>>(pool2, sW1, sb1, sm + SM_SAFH, H_DIM, 384, 1);
  small_mlp<<<B_SZ, 1, 0, stream>>>(sm + SM_SAFH, sW2, sb2, sm + SM_SAFE, 384, 1, 2);
  finalize_output<<<(N_TOK * H_DIM + 255) / 256, 256, 0, stream>>>(
      outAcc, sm + SM_SAFE, dout + O_OUT, dout + O_SAFE);
  avg_conf_kernel<<<1, 256, 0, stream>>>(tconf, dout + O_AVG);

  // ---- evidence head on (scaled) output ----
  launch_gemm(stream, dout + O_OUT, eW1, eb1, bufB, N_TOK, H_DIM, H_DIM, H_DIM, H_DIM, H_DIM, 0, 2, 1.0f);
  launch_gemm(stream, bufB, eW2, eb2, dout + O_EVID, N_TOK, 384, H_DIM, H_DIM, 384, 384, 0, 0, 1.0f);
}
